// AttentionBlock_39084202393711
// MI455X (gfx1250) — matrix operand order repaired and checked by reading, not compiled, not run on hardware
//
#include <hip/hip_runtime.h>

typedef __attribute__((ext_vector_type(16))) _Float16 v16h;
typedef __attribute__((ext_vector_type(8)))  _Float16 v8h;
typedef __attribute__((ext_vector_type(8)))  float    v8f;
typedef int v4i_gcc __attribute__((vector_size(16)));   // matches builtin param type

#define HW 1024
#define C_  512
#define B_  32
#define ATTN_SCALE 0.044194173824159216f  // 1/sqrt(512)

#if defined(__has_builtin)
#if __has_builtin(__builtin_amdgcn_global_load_async_to_lds_b128)
#define HAVE_ASYNC_LDS 1
#endif
#endif

__device__ __forceinline__ void async_copy_b128(const _Float16* gsrc, _Float16* ldst) {
#ifdef HAVE_ASYNC_LDS
  __builtin_amdgcn_global_load_async_to_lds_b128(
      (v4i_gcc*)(const_cast<_Float16*>(gsrc)), (v4i_gcc*)(ldst), 0, 0);
#else
  *(v8h*)ldst = *(const v8h*)gsrc;
#endif
}

__device__ __forceinline__ void wait_async_lds() {
#ifdef HAVE_ASYNC_LDS
#if __has_builtin(__builtin_amdgcn_s_wait_asynccnt)
  __builtin_amdgcn_s_wait_asynccnt(0);
#else
  asm volatile("s_wait_asynccnt 0x0" ::: "memory");
#endif
#endif
}

__device__ __forceinline__ v16h make_v16(v8h lo, v8h hi) {
  union { v16h v; v8h h[2]; } u;
  u.h[0] = lo; u.h[1] = hi;
  return u.v;
}

// A fragment: 16x32 (MxK) f16 from row-major memory, leading dim ld (halfs).
// Per ISA 7.12.2: lane L<16 -> M=L, K=0..7 & 16..23 ; lane L>=16 -> M=L-16, K=8..15 & 24..31.
__device__ __forceinline__ v16h load_a_frag(const _Float16* p, int ld, int lane) {
  int r  = lane & 15;
  int hi = lane >> 4;
  const _Float16* q = p + (size_t)r * ld + 8 * hi;
  v8h lo8 = *(const v8h*)(q);
  v8h hi8 = *(const v8h*)(q + 16);
  return make_v16(lo8, hi8);
}

// B fragment: 32x16 (KxN) f16 from row-major memory (ld in halfs).
// Lane holds column N=lane%16; K=8h..8h+7 & 16+8h..23+8h (h=lane/16).
__device__ __forceinline__ v16h load_b_frag(const _Float16* p, int ld, int lane) {
  int n  = lane & 15;
  int hi = lane >> 4;
  v16h r;
  #pragma unroll
  for (int s = 0; s < 8; s++) {
    r[s]     = p[(size_t)(8 * hi + s) * ld + n];
    r[s + 8] = p[(size_t)(16 + 8 * hi + s) * ld + n];
  }
  return r;
}

__device__ __forceinline__ v8f wmma_f16(v16h a, v16h b, v8f c) {
  return __builtin_amdgcn_wmma_f32_16x16x32_f16(false, a, false, b, (short)0, c, false, false);
}

// ---------------- weight convert + transpose: W[O,C] f32 -> Wt[C,O] f16 --------
__global__ void __launch_bounds__(256) wt_kernel(const float* __restrict__ w,
                                                 _Float16* __restrict__ wt) {
  int i = blockIdx.x * 256 + threadIdx.x;   // 512*512 elements
  int o = i >> 9, c = i & 511;
  wt[(size_t)c * C_ + o] = (_Float16)w[i];
}

// ---------------- GroupNorm: x[B,C,HW] f32 -> h[B,HW,C] f16 --------------------
__global__ void __launch_bounds__(256) gn_kernel(const float* __restrict__ x,
                                                 const float* __restrict__ gnw,
                                                 const float* __restrict__ gnb,
                                                 _Float16* __restrict__ h) {
  int b = blockIdx.x >> 5, g = blockIdx.x & 31;     // 32 groups, 16 ch each
  const float* xp = x + ((size_t)b * C_ + g * 16) * HW;
  float s = 0.f, s2 = 0.f;
  for (int i = threadIdx.x; i < 16 * HW; i += 256) {
    float v = xp[i]; s += v; s2 += v * v;
  }
  __shared__ float rs[256], rq[256];
  rs[threadIdx.x] = s; rq[threadIdx.x] = s2;
  __syncthreads();
  for (int off = 128; off > 0; off >>= 1) {
    if (threadIdx.x < off) {
      rs[threadIdx.x] += rs[threadIdx.x + off];
      rq[threadIdx.x] += rq[threadIdx.x + off];
    }
    __syncthreads();
  }
  float mean = rs[0] * (1.f / 16384.f);
  float var  = rq[0] * (1.f / 16384.f) - mean * mean;
  float inv  = rsqrtf(var + 1e-5f);
  _Float16* hp = h + (size_t)b * HW * C_;
  for (int i = threadIdx.x; i < 16 * HW; i += 256) {
    int cl = i >> 10, pix = i & 1023;
    int c = g * 16 + cl;
    float y = (xp[i] - mean) * inv * gnw[c] + gnb[c];
    hp[(size_t)pix * C_ + c] = (_Float16)y;
  }
}

// ---------------- WMMA GEMM with LDS-staged weight panel ----------------------
#define KBLK 128
#define LDB  136

__global__ void __launch_bounds__(256) gemm_kernel(
    const _Float16* __restrict__ A,   // [B, HW, C]
    const _Float16* __restrict__ Wt,  // [C, O] pre-transposed
    const float* __restrict__ bias,   // [O]
    _Float16* __restrict__ out16,
    float* __restrict__ out32,
    const float* __restrict__ xres,
    int mode) {
  int tid = threadIdx.x;
  int lane = tid & 31, wave = tid >> 5;
  int wr = wave & 3, wc = wave >> 2;
  int b = blockIdx.z;
  int nBlk  = blockIdx.x * 128;
  int mBase = blockIdx.y * 128 + wr * 32;
  int nBase = nBlk + wc * 64;
  const _Float16* Ab = A + ((size_t)b * HW + mBase) * C_;

  __shared__ _Float16 sB[KBLK * LDB];   // 34.8 KB

  v8f acc[2][4] = {};
  for (int ks = 0; ks < C_; ks += KBLK) {
    __syncthreads();   // previous stage fully consumed
    for (int it = tid; it < KBLK * 16; it += 256) {
      int row = it >> 4;
      int c8  = (it & 15) << 3;
      async_copy_b128(Wt + (size_t)(ks + row) * C_ + nBlk + c8, sB + row * LDB + c8);
    }
    if (ks + KBLK < C_)
      __builtin_prefetch(Ab + (size_t)lane * C_ + ks + KBLK, 0, 1);
    wait_async_lds();
    __syncthreads();

    for (int k0 = 0; k0 < KBLK; k0 += 32) {
      v16h a0 = load_a_frag(Ab + ks + k0, C_, lane);
      v16h a1 = load_a_frag(Ab + 16 * C_ + ks + k0, C_, lane);
      const _Float16* bp = sB + (size_t)k0 * LDB + wc * 64;
      v16h b0 = load_b_frag(bp,      LDB, lane);
      v16h b1 = load_b_frag(bp + 16, LDB, lane);
      v16h b2 = load_b_frag(bp + 32, LDB, lane);
      v16h b3 = load_b_frag(bp + 48, LDB, lane);
      acc[0][0] = wmma_f16(a0, b0, acc[0][0]);
      acc[0][1] = wmma_f16(a0, b1, acc[0][1]);
      acc[0][2] = wmma_f16(a0, b2, acc[0][2]);
      acc[0][3] = wmma_f16(a0, b3, acc[0][3]);
      acc[1][0] = wmma_f16(a1, b0, acc[1][0]);
      acc[1][1] = wmma_f16(a1, b1, acc[1][1]);
      acc[1][2] = wmma_f16(a1, b2, acc[1][2]);
      acc[1][3] = wmma_f16(a1, b3, acc[1][3]);
    }
  }

  int hi = lane >> 4, ln = lane & 15;
  for (int ct = 0; ct < 4; ct++) {
    int col = nBase + ct * 16 + ln;
    float bv = bias[col];
    for (int rt = 0; rt < 2; rt++) {
      #pragma unroll
      for (int e = 0; e < 8; e++) {
        int row = mBase + rt * 16 + e + 8 * hi;
        float val = acc[rt][ct][e] + bv;
        if (mode == 0) {
          out16[((size_t)b * HW + row) * C_ + col] = (_Float16)val;
        } else if (mode == 1) {
          out16[((size_t)b * C_ + col) * HW + row] = (_Float16)val;
        } else {
          size_t idx = ((size_t)b * C_ + col) * HW + row;
          out32[idx] = xres[idx] + val;
        }
      }
    }
  }
}

// ---------------- Flash attention: 32 Q-rows per block, online softmax --------
__global__ void __launch_bounds__(256) attn_kernel(
    const _Float16* __restrict__ q,   // [B, HW, C]
    const _Float16* __restrict__ kt,  // [B, C, HW]  (K transposed)
    const _Float16* __restrict__ v,   // [B, HW, C]
    _Float16* __restrict__ aout) {    // [B, HW, C]
  int tid  = threadIdx.x;
  int lane = tid & 31, wave = tid >> 5;
  int hi = lane >> 4, ln = lane & 15;
  int b = blockIdx.y;
  int rBase = blockIdx.x * 32;

  __shared__ float    sS[8 * 32 * 32];
  __shared__ float    sFull[32 * 32];
  __shared__ _Float16 sP[32 * 32];
  __shared__ float    sM[32], sL[32], sScale[32];

  if (tid < 32) { sM[tid] = -1e30f; sL[tid] = 0.f; }
  __syncthreads();

  const _Float16* Qb = q  + ((size_t)b * HW + rBase) * C_;
  const _Float16* Kb = kt + (size_t)b * C_ * HW;
  const _Float16* Vb = v  + (size_t)b * HW * C_;

  int cBase = wave * 64;

  v16h qa[2][2];
  #pragma unroll
  for (int kk = 0; kk < 2; kk++) {
    int c0 = wave * 64 + kk * 32;
    qa[0][kk] = load_a_frag(Qb + c0, C_, lane);
    qa[1][kk] = load_a_frag(Qb + 16 * C_ + c0, C_, lane);
  }

  v8f acc[2][4] = {};

  for (int j = 0; j < 32; j++) {
    int kBase = j * 32;

    v8f sacc[2][2] = {};
    #pragma unroll
    for (int kk = 0; kk < 2; kk++) {
      int c0 = wave * 64 + kk * 32;
      v16h b0 = load_b_frag(Kb + (size_t)c0 * HW + kBase, HW, lane);
      v16h b1 = load_b_frag(Kb + (size_t)c0 * HW + kBase + 16, HW, lane);
      sacc[0][0] = wmma_f16(qa[0][kk], b0, sacc[0][0]);
      sacc[0][1] = wmma_f16(qa[0][kk], b1, sacc[0][1]);
      sacc[1][0] = wmma_f16(qa[1][kk], b0, sacc[1][0]);
      sacc[1][1] = wmma_f16(qa[1][kk], b1, sacc[1][1]);
    }
    float* slab = sS + wave * 1024;
    #pragma unroll
    for (int rt = 0; rt < 2; rt++)
      #pragma unroll
      for (int kt2 = 0; kt2 < 2; kt2++)
        #pragma unroll
        for (int e = 0; e < 8; e++)
          slab[(rt * 16 + e + 8 * hi) * 32 + kt2 * 16 + ln] = sacc[rt][kt2][e];
    __syncthreads();

    for (int idx = tid; idx < 1024; idx += 256) {
      float s = 0.f;
      #pragma unroll
      for (int w8 = 0; w8 < 8; w8++) s += sS[w8 * 1024 + idx];
      sFull[idx] = s * ATTN_SCALE;
    }
    __syncthreads();

    if (tid < 32) {
      int r = tid;
      float m_old = sM[r];
      float rm = -1e30f;
      for (int k = 0; k < 32; k++) rm = fmaxf(rm, sFull[r * 32 + k]);
      float m_new = fmaxf(m_old, rm);
      float es = __expf(m_old - m_new);
      float psum = 0.f;
      for (int k = 0; k < 32; k++) {
        float p = __expf(sFull[r * 32 + k] - m_new);
        psum += p;
        sP[r * 32 + k] = (_Float16)p;
      }
      sL[r] = sL[r] * es + psum;
      sM[r] = m_new;
      sScale[r] = es;
    }
    __syncthreads();

    #pragma unroll
    for (int rt = 0; rt < 2; rt++) {
      float sc[8];
      #pragma unroll
      for (int e = 0; e < 8; e++) sc[e] = sScale[rt * 16 + e + 8 * hi];
      #pragma unroll
      for (int ct = 0; ct < 4; ct++)
        #pragma unroll
        for (int e = 0; e < 8; e++) acc[rt][ct][e] *= sc[e];
    }
    v16h pa0 = load_a_frag((const _Float16*)sP, 32, lane);
    v16h pa1 = load_a_frag((const _Float16*)sP + 16 * 32, 32, lane);
    #pragma unroll
    for (int ct = 0; ct < 4; ct++) {
      v16h vb = load_b_frag(Vb + (size_t)kBase * C_ + cBase + ct * 16, C_, lane);
      acc[0][ct] = wmma_f16(pa0, vb, acc[0][ct]);
      acc[1][ct] = wmma_f16(pa1, vb, acc[1][ct]);
    }
    __syncthreads();
  }

  if (tid < 32) sScale[tid] = 1.0f / sL[tid];
  __syncthreads();
  _Float16* ab = aout + ((size_t)b * HW + rBase) * C_;
  #pragma unroll
  for (int rt = 0; rt < 2; rt++) {
    float inv[8];
    #pragma unroll
    for (int e = 0; e < 8; e++) inv[e] = sScale[rt * 16 + e + 8 * hi];
    #pragma unroll
    for (int ct = 0; ct < 4; ct++) {
      int col = cBase + ct * 16 + ln;
      #pragma unroll
      for (int e = 0; e < 8; e++) {
        int row = rt * 16 + e + 8 * hi;
        ab[(size_t)row * C_ + col] = (_Float16)(acc[rt][ct][e] * inv[e]);
      }
    }
  }
}

extern "C" void kernel_launch(void* const* d_in, const int* in_sizes, int n_in,
                              void* d_out, int out_size, void* d_ws, size_t ws_size,
                              hipStream_t stream) {
  (void)in_sizes; (void)n_in; (void)out_size; (void)ws_size;
  const float* x   = (const float*)d_in[0];
  const float* gnw = (const float*)d_in[1];
  const float* gnb = (const float*)d_in[2];
  const float* wq  = (const float*)d_in[3];
  const float* bq  = (const float*)d_in[4];
  const float* wk  = (const float*)d_in[5];
  const float* bk  = (const float*)d_in[6];
  const float* wv  = (const float*)d_in[7];
  const float* bv  = (const float*)d_in[8];
  const float* wo  = (const float*)d_in[9];
  const float* bo  = (const float*)d_in[10];
  float* out = (float*)d_out;

  size_t NE = (size_t)B_ * HW * C_;
  _Float16* hbuf = (_Float16*)d_ws;
  _Float16* qbuf = hbuf + NE;
  _Float16* ktb  = qbuf + NE;
  _Float16* vbuf = ktb + NE;
  _Float16* abuf = vbuf + NE;
  _Float16* wqt  = abuf + NE;
  _Float16* wkt  = wqt + (size_t)C_ * C_;
  _Float16* wvt  = wkt + (size_t)C_ * C_;
  _Float16* wot  = wvt + (size_t)C_ * C_;

  wt_kernel<<<1024, 256, 0, stream>>>(wq, wqt);
  wt_kernel<<<1024, 256, 0, stream>>>(wk, wkt);
  wt_kernel<<<1024, 256, 0, stream>>>(wv, wvt);
  wt_kernel<<<1024, 256, 0, stream>>>(wo, wot);

  gn_kernel<<<1024, 256, 0, stream>>>(x, gnw, gnb, hbuf);

  dim3 gg(4, 8, 32);
  gemm_kernel<<<gg, 256, 0, stream>>>(hbuf, wqt, bq, qbuf, nullptr, nullptr, 0);
  gemm_kernel<<<gg, 256, 0, stream>>>(hbuf, wkt, bk, ktb,  nullptr, nullptr, 1);
  gemm_kernel<<<gg, 256, 0, stream>>>(hbuf, wvt, bv, vbuf, nullptr, nullptr, 0);

  attn_kernel<<<dim3(32, 32), 256, 0, stream>>>(qbuf, ktb, vbuf, abuf);

  gemm_kernel<<<gg, 256, 0, stream>>>(abuf, wot, bo, nullptr, out, x, 2);
}
